// DPCA2D_30477087932761
// MI455X (gfx1250) — matrix operand order repaired and checked by reading, not compiled, not run on hardware
//
#include <hip/hip_runtime.h>
#include <hip/hip_bf16.h>

typedef __attribute__((ext_vector_type(16))) _Float16 v16h;
typedef __attribute__((ext_vector_type(8)))  float    v8f;

#define B_      4
#define C_      256
#define HW_     4096
#define HEADS_  8
#define DH_     64
#define BH_     32
#define NPTS_   (BH_ * HW_)     // 131072
#define TOPK_   512
#define KMIT_   8

__device__ inline v8f wmma16(v16h a, v16h b, v8f c) {
    return __builtin_amdgcn_wmma_f32_16x16x32_f16(false, a, false, b, (short)0, c,
                                                  false, false);
}

// ---------------------------------------------------------------- f32 -> f16
__global__ void f32_to_f16_kernel(const float* __restrict__ src,
                                  _Float16* __restrict__ dst, int n) {
    int i = blockIdx.x * 256 + threadIdx.x;
    if (i < n) dst[i] = (_Float16)src[i];
}

// ---------------------------------------------------------------- LN stats
__global__ void ln_stats_kernel(const float* __restrict__ ctx, const float* __restrict__ qs,
                                float* meanC, float* rstdC, float* meanQ, float* rstdQ) {
    int tid = blockIdx.x * 256 + threadIdx.x;          // 0 .. 32767
    int which = tid >> 14;
    int pix = tid & 16383;
    const float* src = which ? qs : ctx;
    int b = pix >> 12, p = pix & 4095;
    float s = 0.f, s2 = 0.f;
    for (int c = 0; c < C_; ++c) {
        float v = src[((size_t)b * C_ + c) * HW_ + p];
        s += v; s2 += v * v;
    }
    float mean = s * (1.f / C_);
    float var  = s2 * (1.f / C_) - mean * mean;
    float rstd = rsqrtf(var + 1e-5f);
    if (which) { meanQ[pix] = mean; rstdQ[pix] = rstd; }
    else       { meanC[pix] = mean; rstdC[pix] = rstd; }
}

// ------------------------------------------------- conv1x1 (+LN in, +L2 out)
// out[bh, pixel, dd] f16.  Block tile: M=64 (one head) x N=128 pixels, K=256.
__global__ void __launch_bounds__(256)
qkv_gemm_kernel(const float* __restrict__ X, const float* __restrict__ meanA,
                const float* __restrict__ rstdA, const float* __restrict__ g,
                const float* __restrict__ beta, const _Float16* __restrict__ W,
                _Float16* __restrict__ out, int do_l2) {
    __shared__ __align__(16) _Float16 ldsB[128 * 40];
    __shared__ __align__(16) _Float16 outS[128 * 72];
    int t = threadIdx.x, lane = t & 31, w = t >> 5;
    int lo = lane & 15, hi = lane >> 4;
    int p0 = blockIdx.x * 128, head = blockIdx.y, b = blockIdx.z;
    const float* meanp = meanA + b * HW_ + p0;
    const float* rstdp = rstdA + b * HW_ + p0;
    v8f acc[4] = {};
    for (int kk = 0; kk < C_; kk += 32) {
        // stage normalized B tile  [pixel][k]
        for (int it = 0; it < 16; ++it) {
            int idx = it * 256 + t;
            int pl = idx & 127, kl = idx >> 7;
            float x = X[((size_t)b * C_ + kk + kl) * HW_ + p0 + pl];
            if (kk + 32 < C_)
                __builtin_prefetch(&X[((size_t)b * C_ + kk + 32 + kl) * HW_ + p0 + pl], 0, 0);
            float v = (x - meanp[pl]) * rstdp[pl] * g[kk + kl] + beta[kk + kl];
            ldsB[pl * 40 + kl] = (_Float16)v;
        }
        __syncthreads();
        v16h bf;
        {
            const _Float16* bp = &ldsB[(w * 16 + lo) * 40 + hi * 16];
            #pragma unroll
            for (int h = 0; h < 16; ++h) bf[h] = bp[h];
        }
        #pragma unroll
        for (int f = 0; f < 4; ++f) {
            int row = head * 64 + f * 16 + lo;
            const _Float16* ap = W + (size_t)row * C_ + kk + hi * 16;
            v16h a;
            #pragma unroll
            for (int h = 0; h < 8; ++h) a[h] = ap[h];
            #pragma unroll
            for (int h = 0; h < 8; ++h) a[8 + h] = ap[8 + h];
            acc[f] = wmma16(a, bf, acc[f]);
        }
        __syncthreads();
    }
    float sc = 1.0f;
    if (do_l2) {
        float s = 0.f;
        #pragma unroll
        for (int f = 0; f < 4; ++f)
            #pragma unroll
            for (int r = 0; r < 8; ++r) s += acc[f][r] * acc[f][r];
        s += __shfl_xor(s, 16);
        sc = 1.0f / fmaxf(sqrtf(s), 1e-12f);
    }
    #pragma unroll
    for (int f = 0; f < 4; ++f)
        #pragma unroll
        for (int r = 0; r < 8; ++r) {
            int m = f * 16 + r + hi * 8;
            outS[(w * 16 + lo) * 72 + m] = (_Float16)(acc[f][r] * sc);
        }
    __syncthreads();
    int bh = b * HEADS_ + head;
    {
        int rowl = t >> 1, hb = (t & 1) * 32;
        const uint4* src4 = (const uint4*)&outS[rowl * 72 + hb];
        uint4* dst4 = (uint4*)(out + ((size_t)bh * HW_ + p0 + rowl) * 64 + hb);
        dst4[0] = src4[0]; dst4[1] = src4[1]; dst4[2] = src4[2]; dst4[3] = src4[3];
    }
}

// ---------------------------------------------------------------- k-means
__global__ void zero_kernel(float* p, int n) {
    int i = blockIdx.x * 256 + threadIdx.x;
    if (i < n) p[i] = 0.f;
}

__global__ void init_cent_kernel(const _Float16* __restrict__ qh, float* cent,
                                 _Float16* centh, float* cnormh) {
    int j = blockIdx.x * 256 + threadIdx.x;
    if (j >= TOPK_) return;
    float s = 0.f;
    for (int k = 0; k < DH_; ++k) {
        float c = (float)qh[(size_t)j * DH_ + k];
        cent[j * DH_ + k] = c;
        centh[j * DH_ + k] = (_Float16)c;
        s += c * c;
    }
    cnormh[j] = 0.5f * s;
}

__global__ void update_cent_kernel(float* cent, _Float16* centh, float* cnormh,
                                   const float* __restrict__ sums,
                                   const float* __restrict__ counts) {
    int j = blockIdx.x * 256 + threadIdx.x;
    if (j >= TOPK_) return;
    float cnt = counts[j];
    float s = 0.f;
    for (int k = 0; k < DH_; ++k) {
        float c = cent[j * DH_ + k];
        if (cnt > 0.f) c = sums[j * DH_ + k] / fmaxf(cnt, 1.0f);
        cent[j * DH_ + k] = c;
        centh[j * DH_ + k] = (_Float16)c;
        s += c * c;
    }
    cnormh[j] = 0.5f * s;
}

// block = 16 points x 512 centroids (WMMA), argmax(dot - 0.5|c|^2)
// Point tile (2 KB, contiguous) is brought in once with GLOBAL_LOAD_ASYNC_TO_LDS,
// each wave waits its own ASYNCcnt, then a workgroup barrier publishes the tile.
// mode 0: atomic accumulate sums/counts    mode 1: write L1 dist to center
__global__ void __launch_bounds__(256)
assign_kernel(const _Float16* __restrict__ pts, const _Float16* __restrict__ centh,
              const float* __restrict__ cnormh, const float* __restrict__ cent,
              float* sums, float* counts, float* dist, int mode) {
    int t = threadIdx.x, lane = t & 31, w = t >> 5;
    int lo = lane & 15, hi = lane >> 4;
    size_t P0 = (size_t)blockIdx.x * 16;
    __shared__ __align__(16) _Float16 bstage[16 * DH_];   // [point][k], 2 KB
    __shared__ float redv[8][16];
    __shared__ int   redi[8][16];
    __shared__ int   asg[16];
    // async copy: 256 lanes x 8B == whole 16x64 f16 tile (rows are contiguous)
    {
        unsigned ldsoff = (unsigned)(uintptr_t)(&bstage[0]) + (unsigned)t * 8u;
        const char* gsrc = (const char*)(pts + P0 * DH_) + (size_t)t * 8;
        asm volatile("global_load_async_to_lds_b64 %0, %1, off"
                     :: "v"(ldsoff), "v"(gsrc) : "memory");
        asm volatile("s_wait_asynccnt 0" ::: "memory");
    }
    __syncthreads();
    v8f acc[4] = {};
    #pragma unroll
    for (int kk = 0; kk < DH_; kk += 32) {
        v16h bf;
        const _Float16* bp = &bstage[lo * DH_ + kk + hi * 16];
        #pragma unroll
        for (int h = 0; h < 16; ++h) bf[h] = bp[h];
        #pragma unroll
        for (int f = 0; f < 4; ++f) {
            int m = w * 64 + f * 16 + lo;
            const _Float16* ap = centh + (size_t)m * DH_ + kk + hi * 16;
            v16h a;
            #pragma unroll
            for (int h = 0; h < 8; ++h) a[h] = ap[h];
            #pragma unroll
            for (int h = 0; h < 8; ++h) a[8 + h] = ap[8 + h];
            acc[f] = wmma16(a, bf, acc[f]);
        }
    }
    float best = -3.4e38f; int bidx = 0;
    #pragma unroll
    for (int f = 0; f < 4; ++f)
        #pragma unroll
        for (int r = 0; r < 8; ++r) {
            int m = w * 64 + f * 16 + r + hi * 8;
            float s = acc[f][r] - cnormh[m];
            if (s > best || (s == best && m < bidx)) { best = s; bidx = m; }
        }
    float ov = __shfl_xor(best, 16);
    int   oi = __shfl_xor(bidx, 16);
    if (ov > best || (ov == best && oi < bidx)) { best = ov; bidx = oi; }
    if (lane < 16) { redv[w][lane] = best; redi[w][lane] = bidx; }
    __syncthreads();
    if (t < 16) {
        float bv = redv[0][t]; int bi = redi[0][t];
        for (int ww = 1; ww < 8; ++ww) {
            float v = redv[ww][t]; int i = redi[ww][t];
            if (v > bv || (v == bv && i < bi)) { bv = v; bi = i; }
        }
        asg[t] = bi;
    }
    __syncthreads();
    if (mode == 0) {
        int n = t >> 4, k4 = (t & 15) * 4;
        int a = asg[n];
        const _Float16* pr = &bstage[n * DH_ + k4];
        #pragma unroll
        for (int j = 0; j < 4; ++j)
            atomicAdd(&sums[(size_t)a * DH_ + k4 + j], (float)pr[j]);
        if ((t & 15) == 0) atomicAdd(&counts[a], 1.0f);
    } else {
        if ((t & 15) == 0) {
            int n = t >> 4; int a = asg[n];
            const _Float16* pr = &bstage[n * DH_];
            const float* cr = cent + (size_t)a * DH_;
            float s = 0.f;
            for (int k = 0; k < DH_; ++k) s += fabsf(cr[k] - (float)pr[k]);
            dist[P0 + n] = s;
        }
    }
}

// ---------------------------------------------------------------- top-512
__global__ void topk_kernel(const float* __restrict__ dist, int* best) {
    int bh = blockIdx.x, t = threadIdx.x;
    const float* row = dist + (size_t)bh * HW_;
    __shared__ unsigned hist[256];
    __shared__ unsigned sh_prefix, sh_remaining;
    __shared__ int cnt_strict;
    if (t == 0) { sh_prefix = 0u; sh_remaining = TOPK_; cnt_strict = 0; }
    __syncthreads();
    for (int pass = 0; pass < 4; ++pass) {
        int shift = 24 - pass * 8;
        hist[t] = 0u;
        __syncthreads();
        unsigned pref = sh_prefix;
        for (int i = t; i < HW_; i += 256) {
            unsigned u = __float_as_uint(row[i]);
            bool ok = (pass == 0) || ((u >> (shift + 8)) == (pref >> (shift + 8)));
            if (ok) atomicAdd(&hist[(u >> shift) & 255], 1u);
        }
        __syncthreads();
        if (t == 0) {
            unsigned rem = sh_remaining, accu = 0; int bsel = 0;
            for (int bin = 255; bin >= 0; --bin) {
                unsigned c = hist[bin];
                if (accu + c >= rem) { bsel = bin; rem -= accu; break; }
                accu += c;
            }
            sh_remaining = rem;
            sh_prefix = sh_prefix | ((unsigned)bsel << shift);
        }
        __syncthreads();
    }
    unsigned T = sh_prefix;
    for (int i = t; i < HW_; i += 256) {
        if (__float_as_uint(row[i]) > T) {
            int slot = atomicAdd(&cnt_strict, 1);
            best[bh * TOPK_ + slot] = i;
        }
    }
    __syncthreads();
    if (t == 0) {
        int slot = cnt_strict; unsigned quota = sh_remaining;
        for (int i = 0; i < HW_ && quota > 0; ++i)
            if (__float_as_uint(row[i]) == T) {
                best[bh * TOPK_ + slot++] = i; --quota;
            }
    }
}

// ---------------------------------------------------------------- gather
__global__ void gather_kernel(const _Float16* __restrict__ kh,
                              const _Float16* __restrict__ vh,
                              const int* __restrict__ best,
                              _Float16* kp, _Float16* vpT) {
    int pair = blockIdx.x * 256 + threadIdx.x;     // 0 .. 16383
    int bh = pair >> 9, j = pair & 511;
    int src = best[bh * TOPK_ + j];
    const _Float16* krow = kh + ((size_t)bh * HW_ + src) * DH_;
    const _Float16* vrow = vh + ((size_t)bh * HW_ + src) * DH_;
    _Float16* kdst = kp + ((size_t)bh * TOPK_ + j) * DH_;
    const uint4* ks = (const uint4*)krow;
    uint4* kd = (uint4*)kdst;
    #pragma unroll
    for (int q = 0; q < 8; ++q) kd[q] = ks[q];
    _Float16* vt = vpT + (size_t)bh * DH_ * TOPK_ + j;
    for (int k = 0; k < DH_; ++k) vt[(size_t)k * TOPK_] = vrow[k];
}

// ---------------------------------------------------------------- attention
#define ATTN_SMEM (32 * 520 * 4 + 32 * 520 * 2 + 128)
__global__ void __launch_bounds__(256)
attn_kernel(const _Float16* __restrict__ qh, const _Float16* __restrict__ kp,
            const _Float16* __restrict__ vpT, _Float16* __restrict__ outh) {
    extern __shared__ char smem[];
    float*    S  = (float*)smem;                               // [32][520]
    _Float16* P  = (_Float16*)(smem + 32 * 520 * 4);           // [32][520]
    float*    rs = (float*)(smem + 32 * 520 * 4 + 32 * 520 * 2);
    int t = threadIdx.x, lane = t & 31, w = t >> 5;
    int lo = lane & 15, hi = lane >> 4;
    int bh = blockIdx.y, q0 = blockIdx.x * 32;
    // GEMM1: S = Q x Kp^T   (M=32, N=512, K=64)
    v8f acc1[2][4] = {};
    #pragma unroll
    for (int kk = 0; kk < DH_; kk += 32) {
        v16h a[2];
        #pragma unroll
        for (int f = 0; f < 2; ++f) {
            const _Float16* ap = qh + ((size_t)bh * HW_ + q0 + f * 16 + lo) * DH_ + kk + hi * 16;
            #pragma unroll
            for (int h = 0; h < 8; ++h) a[f][h] = ap[h];
            #pragma unroll
            for (int h = 0; h < 8; ++h) a[f][8 + h] = ap[8 + h];
        }
        #pragma unroll
        for (int s4 = 0; s4 < 4; ++s4) {
            int n = (w + 8 * s4) * 16 + lo;
            const _Float16* bp = kp + ((size_t)bh * TOPK_ + n) * DH_ + kk + hi * 16;
            v16h bf;
            #pragma unroll
            for (int h = 0; h < 16; ++h) bf[h] = bp[h];
            acc1[0][s4] = wmma16(a[0], bf, acc1[0][s4]);
            acc1[1][s4] = wmma16(a[1], bf, acc1[1][s4]);
        }
    }
    #pragma unroll
    for (int f = 0; f < 2; ++f)
        #pragma unroll
        for (int s4 = 0; s4 < 4; ++s4)
            #pragma unroll
            for (int r = 0; r < 8; ++r) {
                int m = f * 16 + r + hi * 8;
                int n = (w + 8 * s4) * 16 + lo;
                S[m * 520 + n] = acc1[f][s4][r];
            }
    __syncthreads();
    // softmax rows (P holds unnormalized exp; 1/sum folded into epilogue)
    for (int i = 0; i < 4; ++i) {
        int m = w * 4 + i;
        float mx = -3.4e38f;
        for (int j = lane; j < TOPK_; j += 32) mx = fmaxf(mx, S[m * 520 + j]);
        #pragma unroll
        for (int o = 16; o >= 1; o >>= 1) mx = fmaxf(mx, __shfl_xor(mx, o));
        float sm = 0.f;
        for (int j = lane; j < TOPK_; j += 32) {
            float e = __expf(S[m * 520 + j] - mx);
            P[m * 520 + j] = (_Float16)e;
            sm += e;
        }
        #pragma unroll
        for (int o = 16; o >= 1; o >>= 1) sm += __shfl_xor(sm, o);
        if (lane == 0) rs[m] = sm;
    }
    __syncthreads();
    // GEMM2: O = P x V   (M=32, N=64, K=512); one fragment per wave
    int f2 = w & 1, s2 = w >> 1;
    v8f acc2 = {};
    for (int kk = 0; kk < TOPK_; kk += 32) {
        v16h a;
        const _Float16* ap = P + (f2 * 16 + lo) * 520 + kk + hi * 16;
        #pragma unroll
        for (int h = 0; h < 8; ++h) a[h] = ap[h];
        #pragma unroll
        for (int h = 0; h < 8; ++h) a[8 + h] = ap[8 + h];
        v16h bf;
        const _Float16* bp = vpT + ((size_t)bh * DH_ + s2 * 16 + lo) * TOPK_ + kk + hi * 16;
        #pragma unroll
        for (int h = 0; h < 16; ++h) bf[h] = bp[h];
        acc2 = wmma16(a, bf, acc2);
    }
    #pragma unroll
    for (int r = 0; r < 8; ++r) {
        int m = f2 * 16 + r + hi * 8;
        float sc = 1.0f / rs[m];
        outh[((size_t)bh * HW_ + q0 + m) * DH_ + s2 * 16 + lo] = (_Float16)(acc2[r] * sc);
    }
}

// ---------------------------------------------------------------- out GEMM
__global__ void __launch_bounds__(256)
out_gemm_kernel(const _Float16* __restrict__ Wout, const _Float16* __restrict__ outh,
                float* __restrict__ y) {
    int t = threadIdx.x, lane = t & 31, w = t >> 5;
    int lo = lane & 15, hi = lane >> 4;
    int p0 = blockIdx.x * 128, mt = blockIdx.y, b = blockIdx.z;
    v8f acc[4] = {};
    for (int kk = 0; kk < 512; kk += 32) {
        int i0 = kk + hi * 16;
        int bh = b * HEADS_ + (i0 >> 6), dd = i0 & 63;
        const _Float16* bp = outh + ((size_t)bh * HW_ + p0 + w * 16 + lo) * DH_ + dd;
        v16h bf;
        #pragma unroll
        for (int h = 0; h < 16; ++h) bf[h] = bp[h];
        #pragma unroll
        for (int f = 0; f < 4; ++f) {
            int row = mt * 64 + f * 16 + lo;
            const _Float16* ap = Wout + (size_t)row * 512 + kk + hi * 16;
            v16h a;
            #pragma unroll
            for (int h = 0; h < 8; ++h) a[h] = ap[h];
            #pragma unroll
            for (int h = 0; h < 8; ++h) a[8 + h] = ap[8 + h];
            acc[f] = wmma16(a, bf, acc[f]);
        }
    }
    #pragma unroll
    for (int f = 0; f < 4; ++f)
        #pragma unroll
        for (int r = 0; r < 8; ++r) {
            int m = mt * 64 + f * 16 + r + hi * 8;
            y[((size_t)b * C_ + m) * HW_ + p0 + w * 16 + lo] = acc[f][r];
        }
}

// ------------------------------------------------------- final LN + residual
__global__ void final_kernel(const float* __restrict__ y, const float* __restrict__ g_out,
                             const float* __restrict__ b_out, const float* __restrict__ gamma,
                             const float* __restrict__ qsrc, float* __restrict__ outF) {
    int pix = blockIdx.x * 256 + threadIdx.x;   // 0 .. 16383
    int b = pix >> 12, p = pix & 4095;
    float s = 0.f, s2 = 0.f;
    for (int o = 0; o < C_; ++o) {
        float v = y[((size_t)b * C_ + o) * HW_ + p];
        s += v; s2 += v * v;
    }
    float mean = s * (1.f / C_);
    float var  = s2 * (1.f / C_) - mean * mean;
    float rstd = rsqrtf(var + 1e-5f);
    float gm = gamma[0];
    for (int o = 0; o < C_; ++o) {
        size_t idx = ((size_t)b * C_ + o) * HW_ + p;
        float v = (y[idx] - mean) * rstd * g_out[o] + b_out[o];
        outF[idx] = gm * v + qsrc[idx];
    }
}

// ================================================================ launch
extern "C" void kernel_launch(void* const* d_in, const int* in_sizes, int n_in,
                              void* d_out, int out_size, void* d_ws, size_t ws_size,
                              hipStream_t stream) {
    (void)in_sizes; (void)n_in; (void)out_size; (void)ws_size;
    const float* qsrc  = (const float*)d_in[0];
    const float* ctx   = (const float*)d_in[1];
    const float* g_ctx = (const float*)d_in[2];
    const float* b_ctx = (const float*)d_in[3];
    const float* g_qs  = (const float*)d_in[4];
    const float* b_qs  = (const float*)d_in[5];
    const float* g_out = (const float*)d_in[6];
    const float* b_out = (const float*)d_in[7];
    const float* w_kv  = (const float*)d_in[8];
    const float* w_q   = (const float*)d_in[9];
    const float* w_out = (const float*)d_in[10];
    const float* gamma = (const float*)d_in[11];
    float* outF = (float*)d_out;

    char* ws = (char*)d_ws;
    size_t off = 0;
    auto carve = [&](size_t bytes) -> char* {
        char* p = ws + off;
        off += (bytes + 255) & ~(size_t)255;
        return p;
    };
    float*    meanC  = (float*)carve(16384 * 4);
    float*    rstdC  = (float*)carve(16384 * 4);
    float*    meanQ  = (float*)carve(16384 * 4);
    float*    rstdQ  = (float*)carve(16384 * 4);
    _Float16* qh     = (_Float16*)carve((size_t)NPTS_ * DH_ * 2);
    _Float16* kh     = (_Float16*)carve((size_t)NPTS_ * DH_ * 2);
    _Float16* vh     = (_Float16*)carve((size_t)NPTS_ * DH_ * 2);
    float*    cent   = (float*)carve(TOPK_ * DH_ * 4);
    _Float16* centh  = (_Float16*)carve(TOPK_ * DH_ * 2);
    float*    cnormh = (float*)carve(TOPK_ * 4);
    float*    sums   = (float*)carve(TOPK_ * DH_ * 4);
    float*    counts = (float*)carve(TOPK_ * 4);
    float*    dist   = (float*)carve((size_t)NPTS_ * 4);
    int*      best   = (int*)carve(BH_ * TOPK_ * 4);
    _Float16* kp     = (_Float16*)carve((size_t)BH_ * TOPK_ * DH_ * 2);
    _Float16* vpT    = (_Float16*)carve((size_t)BH_ * DH_ * TOPK_ * 2);
    _Float16* outh   = (_Float16*)carve((size_t)NPTS_ * DH_ * 2);
    float*    y      = (float*)carve((size_t)B_ * C_ * HW_ * 4);
    _Float16* wqh    = (_Float16*)carve((size_t)512 * C_ * 2);
    _Float16* wkvh   = (_Float16*)carve((size_t)1024 * C_ * 2);
    _Float16* wouth  = (_Float16*)carve((size_t)C_ * 512 * 2);

    // 0. one-time weight conversion to f16 (A operands)
    f32_to_f16_kernel<<<(512 * C_ + 255) / 256, 256, 0, stream>>>(w_q, wqh, 512 * C_);
    f32_to_f16_kernel<<<(1024 * C_ + 255) / 256, 256, 0, stream>>>(w_kv, wkvh, 1024 * C_);
    f32_to_f16_kernel<<<(C_ * 512 + 255) / 256, 256, 0, stream>>>(w_out, wouth, C_ * 512);

    // 1. LN stats for both tensors
    ln_stats_kernel<<<128, 256, 0, stream>>>(ctx, qsrc, meanC, rstdC, meanQ, rstdQ);

    // 2. q/k/v conv1x1 GEMMs (LN fused in, L2-norm fused out for q,k)
    dim3 gQKV(32, HEADS_, B_);
    qkv_gemm_kernel<<<gQKV, 256, 0, stream>>>(qsrc, meanQ, rstdQ, g_qs, b_qs,
                                              wqh, qh, 1);
    qkv_gemm_kernel<<<gQKV, 256, 0, stream>>>(ctx, meanC, rstdC, g_ctx, b_ctx,
                                              wkvh, kh, 1);
    qkv_gemm_kernel<<<gQKV, 256, 0, stream>>>(ctx, meanC, rstdC, g_ctx, b_ctx,
                                              wkvh + (size_t)512 * C_, vh, 0);

    // 3. k-means (8 Lloyd iterations, deterministic init = first 512 q points)
    init_cent_kernel<<<2, 256, 0, stream>>>(qh, cent, centh, cnormh);
    for (int it = 0; it < KMIT_; ++it) {
        zero_kernel<<<(TOPK_ * DH_ + 255) / 256, 256, 0, stream>>>(sums, TOPK_ * DH_);
        zero_kernel<<<(TOPK_ + 255) / 256, 256, 0, stream>>>(counts, TOPK_);
        assign_kernel<<<NPTS_ / 16, 256, 0, stream>>>(qh, centh, cnormh, cent,
                                                      sums, counts, dist, 0);
        update_cent_kernel<<<2, 256, 0, stream>>>(cent, centh, cnormh, sums, counts);
    }
    // final assignment of keys + L1 distance to their center
    assign_kernel<<<NPTS_ / 16, 256, 0, stream>>>(kh, centh, cnormh, cent,
                                                  sums, counts, dist, 1);

    // 4. top-512 farthest keys per (b,head) row
    topk_kernel<<<BH_, 256, 0, stream>>>(dist, best);

    // 5. gather pruned K and transposed V
    gather_kernel<<<BH_ * TOPK_ / 256, 256, 0, stream>>>(kh, vh, best, kp, vpT);

    // 6. attention (QK^T -> softmax -> PV), all WMMA
    hipFuncSetAttribute(reinterpret_cast<const void*>(attn_kernel),
                        hipFuncAttributeMaxDynamicSharedMemorySize, ATTN_SMEM);
    attn_kernel<<<dim3(HW_ / 32, BH_), 256, ATTN_SMEM, stream>>>(qh, kp, vpT, outh);

    // 7. output projection GEMM
    out_gemm_kernel<<<dim3(32, 4, B_), 256, 0, stream>>>(wouth, outh, y);

    // 8. final channel LN + gamma * out + residual
    final_kernel<<<64, 256, 0, stream>>>(y, g_out, b_out, gamma, qsrc, outF);
}
